// FFM_71347996721190
// MI455X (gfx1250) — matrix operand order repaired and checked by reading, not compiled, not run on hardware
//
#include <hip/hip_runtime.h>
#include <hip/hip_bf16.h>
#include <cstdint>

// ---------- types ----------
typedef __attribute__((ext_vector_type(16))) __bf16 v16bf;
typedef __attribute__((ext_vector_type(8)))  float  v8f;
typedef __attribute__((ext_vector_type(4)))  unsigned int v4u;
typedef __attribute__((ext_vector_type(8)))  int v8i;
typedef __attribute__((ext_vector_type(4)))  int v4i;

union FragBF {
    v16bf v;
    v4u   q[2];
};

__device__ __forceinline__ unsigned short f2bf(float f) {
    union { float f; unsigned int u; } x; x.f = f;
    unsigned int u = x.u;
    u += 0x7fffu + ((u >> 16) & 1u);   // round-to-nearest-even
    return (unsigned short)(u >> 16);
}

// ---------- Tensor Data Mover: 2-D bf16 tile -> LDS (padded rows) ----------
// Tile: tileY rows x tileX(=32) elems of 2 bytes, row stride = rowStride elems in
// global memory. LDS padding: after each 64B row, insert 32B -> LDS stride 48 elems.
__device__ __forceinline__ void tdm_load_2d(unsigned lds_byte_off,
                                            const unsigned short* gptr,
                                            unsigned tileX, unsigned tileY,
                                            unsigned long long rowStride)
{
    unsigned long long ga = (unsigned long long)(uintptr_t)gptr;
    v4u g0;
    g0.x = 1u;                                              // count=1 (valid D#)
    g0.y = lds_byte_off;                                    // lds_addr
    g0.z = (unsigned)(ga & 0xFFFFFFFFull);                  // global_addr lo
    g0.w = (unsigned)((ga >> 32) & 0x1FFFFFFull) | (2u << 30); // addr hi + type=2
    v8i g1;
    // data_size=2B(code1), pad_enable=1, pad_interval=16dw(code3), pad_amount=8dw(code7)
    g1[0] = (int)((1u << 16) | (1u << 20) | (3u << 22) | (7u << 25));
    g1[1] = (int)((tileX & 0xFFFFu) << 16);                 // tensor_dim0 lo16
    g1[2] = (int)((tileX >> 16) | ((tileY & 0xFFFFu) << 16)); // td0 hi | td1 lo
    g1[3] = (int)((tileY >> 16) | (tileX << 16));           // td1 hi | tile_dim0
    g1[4] = (int)(tileY & 0xFFFFu);                         // tile_dim1 (tile_dim2=0)
    g1[5] = (int)(rowStride & 0xFFFFFFFFull);               // dim0_stride lo32
    g1[6] = (int)((rowStride >> 32) & 0xFFFFull);           // dim0_stride hi16
    g1[7] = 0;
    v4i z4 = {0, 0, 0, 0};
    v8i z8 = {0, 0, 0, 0, 0, 0, 0, 0};
    __builtin_amdgcn_tensor_load_to_lds(g0, g1, z4, z4, z8, 0);
}

__device__ __forceinline__ unsigned lds_off(const void* p) {
    return (unsigned)(uintptr_t)p;   // flat->LDS: addr[31:0] is the LDS offset
}

// ---------- f32 -> bf16 conversion ----------
__global__ __launch_bounds__(256)
void cvt_bf16_kernel(const float* __restrict__ src, unsigned short* __restrict__ dst, int n)
{
    int i = (blockIdx.x * 256 + threadIdx.x) * 4;
    if (i + 3 < n) {
        #pragma unroll
        for (int j = 0; j < 4; ++j) dst[i + j] = f2bf(src[i + j]);
    }
}

// ---------- big GEMM: C[M,N] = A[M,K] @ W[N,K]^T + bias (A,W bf16; TDM, dbl-buf) ----------
__global__ __launch_bounds__(256)
void gemm_tdm_kernel(const unsigned short* __restrict__ A, const unsigned short* __restrict__ W,
                     const float* __restrict__ bias, float* __restrict__ C,
                     int M, int N, int K)
{
    constexpr int LSTR = 48;
    __shared__ alignas(16) unsigned short ldsA[2][128 * LSTR];   // 2 x 12 KB
    __shared__ alignas(16) unsigned short ldsB[2][128 * LSTR];   // 2 x 12 KB

    const int tid  = threadIdx.x;
    const int lane = tid & 31;
    const int wid  = tid >> 5;
    const int wm   = wid & 1;         // 2 wave-rows of 64
    const int wn   = wid >> 1;        // 4 wave-cols of 32
    const int bm   = blockIdx.y * 128;
    const int bn   = blockIdx.x * 128;

    v8f acc[4][2] = {};

    if (wid == 0) {
        tdm_load_2d(lds_off(&ldsA[0][0]), A + (size_t)bm * K, 32, 128, (unsigned long long)K);
        tdm_load_2d(lds_off(&ldsB[0][0]), W + (size_t)bn * K, 32, 128, (unsigned long long)K);
    }

    const int l16 = lane & 15;
    const int hk  = lane >> 4;
    const int ak  = hk * 8;           // A frag: K {0..7,16..23} / {8..15,24..31}
    const int bk  = hk * 8;           // B frag: K {0..7,16..23} / {8..15,24..31}

    int cur = 0;
    for (int k0 = 0; k0 < K; k0 += 32) {
        if (wid == 0) __builtin_amdgcn_s_wait_tensorcnt(0);
        __syncthreads();                               // buf[cur] ready for all waves
        if (wid == 0 && (k0 + 32) < K) {               // prefetch next tile via TDM
            tdm_load_2d(lds_off(&ldsA[cur ^ 1][0]), A + (size_t)bm * K + k0 + 32, 32, 128, (unsigned long long)K);
            tdm_load_2d(lds_off(&ldsB[cur ^ 1][0]), W + (size_t)bn * K + k0 + 32, 32, 128, (unsigned long long)K);
        }
        const unsigned short* La = &ldsA[cur][0];
        const unsigned short* Lb = &ldsB[cur][0];

        FragBF a[4], b[2];
        #pragma unroll
        for (int i = 0; i < 4; ++i) {
            const int mr = wm * 64 + i * 16 + l16;
            a[i].q[0] = *(const v4u*)&La[mr * LSTR + ak];
            a[i].q[1] = *(const v4u*)&La[mr * LSTR + ak + 16];
        }
        #pragma unroll
        for (int j = 0; j < 2; ++j) {
            const int nr = wn * 32 + j * 16 + l16;
            b[j].q[0] = *(const v4u*)&Lb[nr * LSTR + bk];
            b[j].q[1] = *(const v4u*)&Lb[nr * LSTR + bk + 16];
        }
        #pragma unroll
        for (int i = 0; i < 4; ++i)
            #pragma unroll
            for (int j = 0; j < 2; ++j)
                acc[i][j] = __builtin_amdgcn_wmma_f32_16x16x32_bf16(
                    false, a[i].v, false, b[j].v, (short)0, acc[i][j], false, false);
        __syncthreads();                               // done reading buf[cur]
        cur ^= 1;
    }

    const int hi8 = (lane >> 4) * 8;
    #pragma unroll
    for (int j = 0; j < 2; ++j) {
        const int col = bn + wn * 32 + j * 16 + l16;
        const float bv = bias[col];
        #pragma unroll
        for (int i = 0; i < 4; ++i) {
            const int row0 = bm + wm * 64 + i * 16 + hi8;
            #pragma unroll
            for (int r = 0; r < 8; ++r)
                C[(size_t)(row0 + r) * N + col] = acc[i][j][r] + bv;
        }
    }
}

// ---------- fused small GEMMs: gated = (x@pre_w^T+pre_b) * sigmoid(x@gi_w^T+gi_b) ----------
__global__ __launch_bounds__(256)
void gate_gemm_kernel(const unsigned short* __restrict__ A,
                      const unsigned short* __restrict__ Wp, const unsigned short* __restrict__ Wg,
                      const float* __restrict__ pb, const float* __restrict__ gb,
                      float* __restrict__ gated, int K)
{
    constexpr int LSTR = 48;
    __shared__ alignas(16) unsigned short ldsA[128 * LSTR];
    __shared__ alignas(16) unsigned short ldsP[64 * LSTR];
    __shared__ alignas(16) unsigned short ldsG[64 * LSTR];

    const int tid  = threadIdx.x;
    const int lane = tid & 31;
    const int wid  = tid >> 5;
    const int wm   = wid & 3;         // 4 wave-rows of 32
    const int wn   = wid >> 2;        // 2 wave-cols of 32
    const int bm   = blockIdx.y * 128;

    v8f accP[2][2] = {};
    v8f accG[2][2] = {};

    const int l16 = lane & 15;
    const int hk  = lane >> 4;
    const int ak  = hk * 8;
    const int bk  = hk * 8;

    for (int k0 = 0; k0 < K; k0 += 32) {
        if (wid == 0) {
            tdm_load_2d(lds_off(&ldsA[0]), A + (size_t)bm * K + k0, 32, 128, (unsigned long long)K);
            tdm_load_2d(lds_off(&ldsP[0]), Wp + k0, 32, 64, (unsigned long long)K);
            tdm_load_2d(lds_off(&ldsG[0]), Wg + k0, 32, 64, (unsigned long long)K);
            __builtin_amdgcn_s_wait_tensorcnt(0);
        }
        __syncthreads();

        FragBF a[2], bp[2], bg[2];
        #pragma unroll
        for (int i = 0; i < 2; ++i) {
            const int mr = wm * 32 + i * 16 + l16;
            a[i].q[0] = *(const v4u*)&ldsA[mr * LSTR + ak];
            a[i].q[1] = *(const v4u*)&ldsA[mr * LSTR + ak + 16];
        }
        #pragma unroll
        for (int j = 0; j < 2; ++j) {
            const int nr = wn * 32 + j * 16 + l16;
            bp[j].q[0] = *(const v4u*)&ldsP[nr * LSTR + bk];
            bp[j].q[1] = *(const v4u*)&ldsP[nr * LSTR + bk + 16];
            bg[j].q[0] = *(const v4u*)&ldsG[nr * LSTR + bk];
            bg[j].q[1] = *(const v4u*)&ldsG[nr * LSTR + bk + 16];
        }
        #pragma unroll
        for (int i = 0; i < 2; ++i)
            #pragma unroll
            for (int j = 0; j < 2; ++j) {
                accP[i][j] = __builtin_amdgcn_wmma_f32_16x16x32_bf16(
                    false, a[i].v, false, bp[j].v, (short)0, accP[i][j], false, false);
                accG[i][j] = __builtin_amdgcn_wmma_f32_16x16x32_bf16(
                    false, a[i].v, false, bg[j].v, (short)0, accG[i][j], false, false);
            }
        __syncthreads();
    }

    const int hi8 = (lane >> 4) * 8;
    #pragma unroll
    for (int j = 0; j < 2; ++j) {
        const int col = wn * 32 + j * 16 + l16;
        const float pbv = pb[col], gbv = gb[col];
        #pragma unroll
        for (int i = 0; i < 2; ++i) {
            const int row0 = bm + wm * 32 + i * 16 + hi8;
            #pragma unroll
            for (int r = 0; r < 8; ++r) {
                float pre = accP[i][j][r] + pbv;
                float g   = 1.0f / (1.0f + expf(-(accG[i][j][r] + gbv)));
                gated[(size_t)(row0 + r) * 64 + col] = pre * g;
            }
        }
    }
}

// ---------- resettable complex scan: one wave per (trace,ctx) channel ----------
__global__ __launch_bounds__(32)
void scan_kernel(const float* __restrict__ gated,
                 const float* __restrict__ st_re, const float* __restrict__ st_im,
                 const unsigned char* __restrict__ start,
                 const float* __restrict__ ffa_a, const float* __restrict__ ffa_b,
                 unsigned short* __restrict__ zin,   // (8192 x 2048) bf16
                 float* __restrict__ fs)             // final state (re,im interleaved)
{
    const int ch = blockIdx.x;       // 0..1023
    const int m  = ch >> 4;
    const int c  = ch & 15;
    const int lane = threadIdx.x;    // wave32

    const float decay = expf(-fabsf(ffa_a[m]));
    const float gr = decay * cosf(ffa_b[c]);
    const float gi = decay * sinf(ffa_b[c]);

    constexpr int SEG = 8192 / 32;
    const int t0 = lane * SEG;

    // pass 1: segment-local affine map (A,B): s_out = A*s_in + B
    float aR = 1.f, aI = 0.f, bR = 0.f, bI = 0.f;
    for (int i = 0; i < SEG; ++i) {
        const int t = t0 + i;
        const float x = gated[t * 64 + m];
        if (start[t]) { aR = 0.f; aI = 0.f; bR = x; bI = 0.f; }
        else {
            float nAR = aR * gr - aI * gi, nAI = aR * gi + aI * gr;
            float nBR = bR * gr - bI * gi + x, nBI = bR * gi + bI * gr;
            aR = nAR; aI = nAI; bR = nBR; bI = nBI;
        }
    }
    // inclusive scan of maps across lanes
    for (int off = 1; off < 32; off <<= 1) {
        float paR = __shfl_up(aR, off), paI = __shfl_up(aI, off);
        float pbR = __shfl_up(bR, off), pbI = __shfl_up(bI, off);
        if (lane >= off) {
            float nbR = aR * pbR - aI * pbI + bR;
            float nbI = aR * pbI + aI * pbR + bI;
            float naR = aR * paR - aI * paI;
            float naI = aR * paI + aI * paR;
            aR = naR; aI = naI; bR = nbR; bI = nbI;
        }
    }
    float eaR = __shfl_up(aR, 1), eaI = __shfl_up(aI, 1);
    float ebR = __shfl_up(bR, 1), ebI = __shfl_up(bI, 1);
    if (lane == 0) { eaR = 1.f; eaI = 0.f; ebR = 0.f; ebI = 0.f; }
    const float s0R = st_re[ch], s0I = st_im[ch];
    float sR = eaR * s0R - eaI * s0I + ebR;
    float sI = eaR * s0I + eaI * s0R + ebI;

    // pass 2: emit all states as bf16
    for (int i = 0; i < SEG; ++i) {
        const int t = t0 + i;
        const float x = gated[t * 64 + m];
        if (start[t]) { sR = x; sI = 0.f; }
        else {
            float nR = sR * gr - sI * gi + x;
            float nI = sR * gi + sI * gr;
            sR = nR; sI = nI;
        }
        zin[(size_t)t * 2048 + ch]        = f2bf(sR);
        zin[(size_t)t * 2048 + 1024 + ch] = f2bf(sI);
    }
    if (lane == 31) { fs[2 * ch] = sR; fs[2 * ch + 1] = sI; }
}

// ---------- layernorm + gated skip, one block per row ----------
__global__ __launch_bounds__(256)
void ln_kernel(const float* __restrict__ z, const float* __restrict__ go,
               const float* __restrict__ sk, float* __restrict__ out)
{
    const int t = blockIdx.x;
    const int tid = threadIdx.x;
    float zg[4], st[4];
    float sum = 0.f, sq = 0.f;
    #pragma unroll
    for (int j = 0; j < 4; ++j) {
        const int col = tid + j * 256;
        const size_t idx = (size_t)t * 1024 + col;
        float g = 1.0f / (1.0f + expf(-go[idx]));
        float v = z[idx] * g;
        zg[j] = v;
        st[j] = sk[idx] * (1.0f - g);
        sum += v; sq += v * v;
    }
    __shared__ float red[16];
    for (int off = 16; off; off >>= 1) {
        sum += __shfl_down(sum, off);
        sq  += __shfl_down(sq, off);
    }
    const int lane = tid & 31, wid = tid >> 5;
    if (lane == 0) { red[wid] = sum; red[8 + wid] = sq; }
    __syncthreads();
    if (tid == 0) {
        float s = 0.f, q = 0.f;
        for (int w = 0; w < 8; ++w) { s += red[w]; q += red[8 + w]; }
        red[0] = s; red[1] = q;
    }
    __syncthreads();
    const float mu  = red[0] * (1.0f / 1024.0f);
    const float var = red[1] * (1.0f / 1024.0f) - mu * mu;
    const float inv = rsqrtf(var + 1e-5f);
    #pragma unroll
    for (int j = 0; j < 4; ++j) {
        const int col = tid + j * 256;
        out[(size_t)t * 1024 + col] = (zg[j] - mu) * inv + st[j];
    }
}

// ---------- launcher ----------
extern "C" void kernel_launch(void* const* d_in, const int* in_sizes, int n_in,
                              void* d_out, int out_size, void* d_ws, size_t ws_size,
                              hipStream_t stream)
{
    constexpr int T = 8192, IN = 1024, TR = 64, CX = 16, OUT = 1024;

    const float* x      = (const float*)d_in[0];
    const float* st_re  = (const float*)d_in[1];
    const float* st_im  = (const float*)d_in[2];
    const unsigned char* start = (const unsigned char*)d_in[3];
    const float* pre_w  = (const float*)d_in[5];
    const float* pre_b  = (const float*)d_in[6];
    const float* gi_w   = (const float*)d_in[7];
    const float* gi_b   = (const float*)d_in[8];
    const float* go_w   = (const float*)d_in[9];
    const float* go_b   = (const float*)d_in[10];
    const float* skip_w = (const float*)d_in[11];
    const float* skip_b = (const float*)d_in[12];
    const float* mix_w  = (const float*)d_in[13];
    const float* mix_b  = (const float*)d_in[14];
    const float* ffa_a  = (const float*)d_in[15];
    const float* ffa_b  = (const float*)d_in[16];

    float* out = (float*)d_out;
    float* fs  = out + ((size_t)out_size - 2 * TR * CX);

    // workspace layout
    char* base = (char*)d_ws;
    size_t o = 0;
    unsigned short* xb   = (unsigned short*)(base + o); o += (size_t)T * IN * 2;        // 16 MB
    unsigned short* wpre = (unsigned short*)(base + o); o += (size_t)TR * IN * 2;       // 128 KB
    unsigned short* wgi  = (unsigned short*)(base + o); o += (size_t)TR * IN * 2;       // 128 KB
    unsigned short* wgo  = (unsigned short*)(base + o); o += (size_t)OUT * IN * 2;      // 2 MB
    unsigned short* wsk  = (unsigned short*)(base + o); o += (size_t)OUT * IN * 2;      // 2 MB
    unsigned short* wmix = (unsigned short*)(base + o); o += (size_t)OUT * 2 * IN * 2;  // 4 MB
    float* gated    = (float*)(base + o); o += (size_t)T * TR * 4;                      // 2 MB
    unsigned short* zin = (unsigned short*)(base + o); o += (size_t)T * 2 * TR * CX * 2;// 32 MB
    float* z_lin    = (float*)(base + o); o += (size_t)T * OUT * 4;                     // 32 MB
    float* go_lin   = (float*)(base + o); o += (size_t)T * OUT * 4;                     // 32 MB
    float* skip_lin = (float*)(base + o); o += (size_t)T * OUT * 4;                     // 32 MB
    (void)ws_size; (void)n_in; (void)in_sizes;

    const dim3 blk(256);

    // one-time f32 -> bf16 conversions (x read once; GEMMs then move pure bf16 via TDM)
    cvt_bf16_kernel<<<(T * IN) / 1024, blk, 0, stream>>>(x, xb, T * IN);
    cvt_bf16_kernel<<<(TR * IN) / 1024, blk, 0, stream>>>(pre_w, wpre, TR * IN);
    cvt_bf16_kernel<<<(TR * IN) / 1024, blk, 0, stream>>>(gi_w, wgi, TR * IN);
    cvt_bf16_kernel<<<(OUT * IN) / 1024, blk, 0, stream>>>(go_w, wgo, OUT * IN);
    cvt_bf16_kernel<<<(OUT * IN) / 1024, blk, 0, stream>>>(skip_w, wsk, OUT * IN);
    cvt_bf16_kernel<<<(OUT * 2 * IN) / 1024, blk, 0, stream>>>(mix_w, wmix, OUT * 2 * IN);

    // fused pre/gate_in GEMM -> gated_x
    gate_gemm_kernel<<<dim3(1, T / 128), blk, 0, stream>>>(xb, wpre, wgi, pre_b, gi_b, gated, IN);

    // parallel resettable complex scan -> z_in (bf16) + final_state
    scan_kernel<<<TR * CX, 32, 0, stream>>>(gated, st_re, st_im, start, ffa_a, ffa_b, zin, fs);

    // big GEMMs (TDM double-buffered, 128x128 tiles)
    gemm_tdm_kernel<<<dim3(OUT / 128, T / 128), blk, 0, stream>>>(zin, wmix, mix_b, z_lin, T, OUT, 2 * TR * CX);
    gemm_tdm_kernel<<<dim3(OUT / 128, T / 128), blk, 0, stream>>>(xb, wgo, go_b, go_lin, T, OUT, IN);
    gemm_tdm_kernel<<<dim3(OUT / 128, T / 128), blk, 0, stream>>>(xb, wsk, skip_b, skip_lin, T, OUT, IN);

    // layernorm + gated skip
    ln_kernel<<<T, blk, 0, stream>>>(z_lin, go_lin, skip_lin, out);
}
